// CausalLinearAttention_17171279249472
// MI455X (gfx1250) — compile-verified, matrix operand order checked by reading, not run on hardware
//
#include <hip/hip_runtime.h>
#include <stdint.h>

// Causal linear attention (elu+1 feature map), chunked recurrence, f16 WMMA,
// async-to-LDS double staging. One workgroup per (n,h); 8 waves; chunk C=128;
// V padded with a ones-column so cumsum(K) rides along as state column 64.
// State master lives in per-wave WMMA accumulator registers (f32); an f16
// shadow in LDS feeds the inter-chunk B operands.

#define NB  4
#define LL  8192
#define HH  16
#define EE  64
#define CC  128
#define NCH (LL / CC)
#define DP  80   // padded value width: 64 data + 1 ones + 15 zeros (5 tiles)

typedef __attribute__((ext_vector_type(16))) _Float16 v16h;
typedef __attribute__((ext_vector_type(8)))  _Float16 v8h;
typedef __attribute__((ext_vector_type(8)))  float    v8f;

// ---- LDS layout (byte offsets inside dynamic smem), all 16B aligned ----
#define OFF_QH   0u                          // [128][64] f16 featmap(Q)
#define OFF_KH   (OFF_QH  + CC*EE*2u)        // [128][64] f16 featmap(K)
#define OFF_KT   (OFF_KH  + CC*EE*2u)        // [64][128] f16 featmap(K)^T
#define OFF_VT   (OFF_KT  + EE*CC*2u)        // [80][128] f16 V'^T
#define OFF_SC   (OFF_VT  + DP*CC*2u)        // [128][128] f16 masked scores
#define OFF_STH  (OFF_SC  + CC*CC*2u)        // [80][64]  f16 state'^T shadow
#define OFF_DNM  (OFF_STH + DP*EE*2u)        // [128]     f32 normalizer
#define OFF_RAWQ (OFF_DNM + CC*4u)           // [128][64] f32 async landing
#define OFF_RAWK (OFF_RAWQ + CC*EE*4u)
#define OFF_RAWV (OFF_RAWK + CC*EE*4u)
#define SMEM_BYTES (OFF_RAWV + CC*EE*4u)     // ~207 KB

__device__ __forceinline__ float featmap(float x) {
  return x > 0.f ? x + 1.f : __expf(x);   // elu(x)+1
}

__device__ __forceinline__ v8f wmma16(v16h a, v16h b, v8f c) {
  return __builtin_amdgcn_wmma_f32_16x16x32_f16(
      false, a, false, b, (short)0, c, false, false);
}

// 16x32 f16 operand fragment from a row-major f16 LDS array.
// gfx1250 16-bit operand layout: lane L holds row (L&15); halves 0..7 carry
// K = kb..kb+7, halves 8..15 carry K = 16+kb..16+kb+7, kb = (L>>4)*8.
__device__ __forceinline__ v16h frag16(const _Float16* X, int row, int stride,
                                       int k0, int kb) {
  const _Float16* p = X + row * stride + k0 + kb;
  v8h lo = *(const v8h*)p;
  v8h hi = *(const v8h*)(p + 16);
  v16h r;
#pragma unroll
  for (int j = 0; j < 8; ++j) { r[j] = lo[j]; r[j + 8] = hi[j]; }
  return r;
}

// CDNA5 async memory: per-lane 16B global -> LDS, tracked by ASYNCcnt.
__device__ __forceinline__ void async_b128(unsigned lds_off, const float* g) {
  asm volatile("global_load_async_to_lds_b128 %0, %1, off"
               :: "v"(lds_off), "v"((uint64_t)(uintptr_t)g) : "memory");
}
__device__ __forceinline__ void wait_async() {
  asm volatile("s_wait_asynccnt 0" ::: "memory");
}

__device__ __forceinline__ void issue_chunk_loads(
    const float* Qg, const float* Kg, const float* Vg,
    int n, int h, int ch, int rowp, int c0,
    unsigned ldsQ, unsigned ldsK, unsigned ldsV) {
  size_t base = (((size_t)n * LL + (size_t)ch * CC + rowp) * HH + h) * EE + c0;
  unsigned loff = (unsigned)((rowp * EE + c0) * 4);
#pragma unroll
  for (int g = 0; g < 8; ++g) {
    async_b128(ldsQ + loff + g * 16u, Qg + base + g * 4);
    async_b128(ldsK + loff + g * 16u, Kg + base + g * 4);
    async_b128(ldsV + loff + g * 16u, Vg + base + g * 4);
  }
}

extern "C" __global__ __launch_bounds__(256, 1)
void cla_chunked_wmma(const float* __restrict__ Qg, const float* __restrict__ Kg,
                      const float* __restrict__ Vg, float* __restrict__ Og) {
  extern __shared__ char smem_raw[];
  _Float16* Qh  = (_Float16*)(smem_raw + OFF_QH);
  _Float16* Kh  = (_Float16*)(smem_raw + OFF_KH);
  _Float16* Kt  = (_Float16*)(smem_raw + OFF_KT);
  _Float16* Vt  = (_Float16*)(smem_raw + OFF_VT);
  _Float16* Sc  = (_Float16*)(smem_raw + OFF_SC);
  _Float16* stH = (_Float16*)(smem_raw + OFF_STH);
  float*    dnm = (float*)   (smem_raw + OFF_DNM);
  float*    rawQ = (float*)  (smem_raw + OFF_RAWQ);
  float*    rawK = (float*)  (smem_raw + OFF_RAWK);
  float*    rawV = (float*)  (smem_raw + OFF_RAWV);
  // LDS byte offsets for the async engine (flat addr truncates to LDS offset)
  const unsigned lds0 = (unsigned)(uintptr_t)smem_raw;
  const unsigned ldsQ = lds0 + OFF_RAWQ, ldsK = lds0 + OFF_RAWK,
                 ldsV = lds0 + OFF_RAWV;

  const int tid  = threadIdx.x;
  const int lane = tid & 31;
  const int w    = tid >> 5;      // wave id == row tile (0..7)
  const int m16  = lane & 15;
  const int hi   = lane >> 4;
  const int kb   = hi * 8;

  const int n = blockIdx.x >> 4;
  const int h = blockIdx.x & 15;

  const int rowp = tid >> 1;        // loader: 2 threads per chunk row
  const int c0   = (tid & 1) * 32;  // each covers 32 of the 64 features
  const size_t rowstride = (size_t)HH * EE;

  // ---- prologue: start chunk-0 staging, then init LDS while it flies
  issue_chunk_loads(Qg, Kg, Vg, n, h, 0, rowp, c0, ldsQ, ldsK, ldsV);
  for (int i = tid; i < DP * EE; i += 256) stH[i] = (_Float16)0.f;
  for (int i = tid; i < CC * CC; i += 256) Sc[i] = (_Float16)0.f;
  for (int i = tid; i < 16 * CC; i += 256) {
    int d = 64 + (i >> 7), m = i & 127;
    Vt[d * CC + m] = (_Float16)((d == 64) ? 1.f : 0.f);
  }
  // per-wave persistent f32 state tiles: t = w, w+8, w+16 (4 e-tiles x 5 d-tiles)
  v8f st[3];
#pragma unroll
  for (int u = 0; u < 3; ++u)
    st[u] = (v8f){0.f, 0.f, 0.f, 0.f, 0.f, 0.f, 0.f, 0.f};

  wait_async();
  __syncthreads();

  for (int ch = 0; ch < NCH; ++ch) {
    // ========== Phase A: raw LDS f32 -> featmap -> f16 operand arrays ======
    {
      const float4* rq = (const float4*)(rawQ + rowp * EE + c0);
      const float4* rk = (const float4*)(rawK + rowp * EE + c0);
      const float4* rv = (const float4*)(rawV + rowp * EE + c0);
      _Float16 qh[32], kh[32], vh[32];
#pragma unroll
      for (int g = 0; g < 8; ++g) {
        float4 q4 = rq[g], k4 = rk[g], v4 = rv[g];
        qh[4*g+0] = (_Float16)featmap(q4.x); qh[4*g+1] = (_Float16)featmap(q4.y);
        qh[4*g+2] = (_Float16)featmap(q4.z); qh[4*g+3] = (_Float16)featmap(q4.w);
        kh[4*g+0] = (_Float16)featmap(k4.x); kh[4*g+1] = (_Float16)featmap(k4.y);
        kh[4*g+2] = (_Float16)featmap(k4.z); kh[4*g+3] = (_Float16)featmap(k4.w);
        vh[4*g+0] = (_Float16)v4.x; vh[4*g+1] = (_Float16)v4.y;
        vh[4*g+2] = (_Float16)v4.z; vh[4*g+3] = (_Float16)v4.w;
      }
#pragma unroll
      for (int g = 0; g < 4; ++g) {
        v8h tq, tk;
#pragma unroll
        for (int j = 0; j < 8; ++j) { tq[j] = qh[g*8+j]; tk[j] = kh[g*8+j]; }
        *(v8h*)(Qh + rowp * EE + c0 + g * 8) = tq;   // ds_store_b128
        *(v8h*)(Kh + rowp * EE + c0 + g * 8) = tk;
      }
#pragma unroll
      for (int j = 0; j < 32; ++j) {                 // transposed copies
        Kt[(c0 + j) * CC + rowp] = kh[j];
        Vt[(c0 + j) * CC + rowp] = vh[j];
      }
    }
    __syncthreads();   // raw consumed + f16 arrays ready

    // kick off next chunk's staging; overlaps with phases B..D below
    if (ch + 1 < NCH)
      issue_chunk_loads(Qg, Kg, Vg, n, h, ch + 1, rowp, c0, ldsQ, ldsK, ldsV);

    // ========== Phase B: masked score tiles for row tile w =================
    v16h qa0 = frag16(Qh, w * 16 + m16, EE, 0,  kb);
    v16h qa1 = frag16(Qh, w * 16 + m16, EE, 32, kb);
    for (int c = 0; c <= w; ++c) {
      v8f s = {0.f, 0.f, 0.f, 0.f, 0.f, 0.f, 0.f, 0.f};
      v16h b0 = frag16(Kh, c * 16 + m16, EE, 0,  kb);  // B[e][col]=K[col][e]
      v16h b1 = frag16(Kh, c * 16 + m16, EE, 32, kb);
      s = wmma16(qa0, b0, s);
      s = wmma16(qa1, b1, s);
      if (c == w) {                       // tril mask on the diagonal tile
#pragma unroll
        for (int i = 0; i < 8; ++i)
          if (m16 > i + hi * 8) s[i] = 0.f;
      }
#pragma unroll
      for (int i = 0; i < 8; ++i)        // C layout: row=i+hi*8, col=m16
        Sc[(w * 16 + hi * 8 + i) * CC + c * 16 + m16] = (_Float16)s[i];
    }
    // (no barrier: each wave re-reads only its own score rows)

    // ========== Phase C: O = Q·state' + tril(QK^T)·V' ======================
    v8f acc[5];
#pragma unroll
    for (int ct = 0; ct < 5; ++ct)
      acc[ct] = (v8f){0.f, 0.f, 0.f, 0.f, 0.f, 0.f, 0.f, 0.f};
#pragma unroll
    for (int ct = 0; ct < 5; ++ct) {     // inter: Q (16x64) @ state' (64x80)
      v16h b0 = frag16(stH, ct * 16 + m16, EE, 0,  kb);
      v16h b1 = frag16(stH, ct * 16 + m16, EE, 32, kb);
      acc[ct] = wmma16(qa0, b0, acc[ct]);
      acc[ct] = wmma16(qa1, b1, acc[ct]);
    }
    const int nst = w / 2 + 1;           // k-steps at/below the diagonal
    for (int s = 0; s < nst; ++s) {
      v16h sa = frag16(Sc, w * 16 + m16, CC, s * 32, kb);
#pragma unroll
      for (int ct = 0; ct < 5; ++ct) {
        v16h vb = frag16(Vt, ct * 16 + m16, CC, s * 32, kb);
        acc[ct] = wmma16(sa, vb, acc[ct]);
      }
    }

    // normalizer = O' column 64 (tile 4, in-tile col 0 -> lanes 0 and 16)
    if (m16 == 0) {
#pragma unroll
      for (int i = 0; i < 8; ++i) dnm[w * 16 + hi * 8 + i] = acc[4][i];
    }
    __syncthreads();   // also fences stH reads (C) from stH writes (D)

    // ========== Phase E: normalize + store =================================
    {
      float rz[8];
#pragma unroll
      for (int i = 0; i < 8; ++i)
        rz[i] = 1.f / (dnm[w * 16 + hi * 8 + i] + 1e-6f);
      size_t obase =
          (((size_t)n * LL + (size_t)ch * CC + w * 16 + hi * 8) * HH + h) * EE + m16;
#pragma unroll
      for (int ct = 0; ct < 4; ++ct)
#pragma unroll
        for (int i = 0; i < 8; ++i)
          Og[obase + (size_t)i * rowstride + ct * 16] = acc[ct][i] * rz[i];
    }

    // ========== Phase D: state' += K^T @ V' (register-resident f32) ========
#pragma unroll
    for (int u = 0; u < 3; ++u) {
      int t = w + u * 8;                 // tile id: et = t&3, dt = t>>2
      if (t < 20) {
        int et = t & 3, dt = t >> 2;
        v8f c = st[u];
#pragma unroll
        for (int s = 0; s < 4; ++s) {
          v16h ka = frag16(Kt, et * 16 + m16, CC, s * 32, kb); // A[e][m]=K[m][e]
          v16h vb = frag16(Vt, dt * 16 + m16, CC, s * 32, kb); // B[m][d]=V'[m][d]
          c = wmma16(ka, vb, c);
        }
        st[u] = c;
        v8h hc;                          // f16 shadow for next chunk's Phase C
#pragma unroll
        for (int i = 0; i < 8; ++i) hc[i] = (_Float16)c[i];
        *(v8h*)(stH + (dt * 16 + m16) * EE + et * 16 + hi * 8) = hc;
      }
    }

    wait_async();      // next chunk's raw staging complete (per-wave)
    __syncthreads();   // all waves' staging + state shadow visible
  }
}

extern "C" void kernel_launch(void* const* d_in, const int* in_sizes, int n_in,
                              void* d_out, int out_size, void* d_ws, size_t ws_size,
                              hipStream_t stream) {
  (void)in_sizes; (void)n_in; (void)out_size; (void)d_ws; (void)ws_size;
  const float* Q = (const float*)d_in[0];
  const float* K = (const float*)d_in[1];
  const float* V = (const float*)d_in[2];
  float* O = (float*)d_out;

  hipFuncSetAttribute((const void*)cla_chunked_wmma,
                      hipFuncAttributeMaxDynamicSharedMemorySize,
                      (int)SMEM_BYTES);
  cla_chunked_wmma<<<dim3(NB * HH), dim3(256), SMEM_BYTES, stream>>>(Q, K, V, O);
}
